// SequenceAttention_3693671874659
// MI455X (gfx1250) — matrix operand order repaired and checked by reading, not compiled, not run on hardware
//
#include <hip/hip_runtime.h>
#include <hip/hip_bf16.h>
#include <math.h>

// Problem constants (B,S,D,H) = (4, 2048, 2048, 16), HD = 128.
#define B_   4
#define S_   2048
#define D_   2048
#define H_   16
#define HD_  128
#define NT_  (B_ * S_)      // 8192 tokens
#define QKVN (3 * D_)       // 6144

typedef __attribute__((ext_vector_type(16))) __bf16 v16bf;
typedef __attribute__((ext_vector_type(8)))  __bf16 v8bf;
typedef __attribute__((ext_vector_type(8)))  float  v8f;

// ln(10000)/128 for RoPE inv_freq = exp(-(2i)*ln(theta)/HD)
#define ROPE_LN_OVER_HD (9.210340371976184f / 128.0f)

__device__ __forceinline__ v16bf cvt16(v8f lo, v8f hi) {
    v16bf r;
#pragma unroll
    for (int i = 0; i < 8; ++i) { r[i] = (__bf16)lo[i]; r[8 + i] = (__bf16)hi[i]; }
    return r;
}
__device__ __forceinline__ v16bf pack16(v8bf lo, v8bf hi) {
    v16bf r;
#pragma unroll
    for (int i = 0; i < 8; ++i) { r[i] = lo[i]; r[8 + i] = hi[i]; }
    return r;
}

// ---------------------------------------------------------------------------
// Kernel 1: qkv = x @ Wqkv^T + bqkv, fused RoPE(q,k), pack bf16.
//   Q,K -> [B,H,S,HD] (rope applied), V -> [B,H,HD,S] (transposed for P@V).
// One wave computes a 32x32 output block: 4 WMMAs per K-step sharing
// 2 A-fragment + 2 B-fragment loads (all 32-byte vector loads).
// ---------------------------------------------------------------------------
__global__ __launch_bounds__(256)
void qkv_rope_kernel(const float* __restrict__ x,
                     const float* __restrict__ Wqkv,
                     const float* __restrict__ bqkv,
                     __bf16* __restrict__ qb,
                     __bf16* __restrict__ kb,
                     __bf16* __restrict__ vtb)
{
    const int wave = blockIdx.x * 8 + (threadIdx.x >> 5);
    const int lane = threadIdx.x & 31;
    const int mt = wave % (NT_ / 32);
    const int nt = wave / (NT_ / 32);
    if (nt >= QKVN / 32) return;

    const int row0  = mt * 32;
    const int col0  = nt * 32;
    const int arow  = lane & 15;                // A-matrix row (M)
    const int aklo  = (lane < 16) ? 0 : 16;     // A-matrix K sub-offset
    const int bcol  = lane & 15;                // B/C column (N)
    const int bkoff = (lane < 16) ? 0 : 16;     // B-matrix K offset
    const int mhalf = (lane < 16) ? 0 : 8;      // C rows: vgpr j -> M = mhalf + j

    const float* __restrict__ xr0 = x + (size_t)(row0 + arow) * D_;
    const float* __restrict__ xr1 = xr0 + (size_t)16 * D_;
    const float* __restrict__ wr0 = Wqkv + (size_t)(col0 + bcol) * D_;
    const float* __restrict__ wr1 = wr0 + (size_t)16 * D_;

    const v8f vzero = {};
    v8f acc[2][2];
    acc[0][0] = vzero; acc[0][1] = vzero; acc[1][0] = vzero; acc[1][1] = vzero;

    for (int k0 = 0; k0 < D_; k0 += 32) {
        v16bf a0 = cvt16(*(const v8f*)(xr0 + k0 + aklo),
                         *(const v8f*)(xr0 + k0 + 8 + aklo));
        v16bf a1 = cvt16(*(const v8f*)(xr1 + k0 + aklo),
                         *(const v8f*)(xr1 + k0 + 8 + aklo));
        v16bf b0 = cvt16(*(const v8f*)(wr0 + k0 + bkoff),
                         *(const v8f*)(wr0 + k0 + bkoff + 8));
        v16bf b1 = cvt16(*(const v8f*)(wr1 + k0 + bkoff),
                         *(const v8f*)(wr1 + k0 + bkoff + 8));
        acc[0][0] = __builtin_amdgcn_wmma_f32_16x16x32_bf16(false, a0, false, b0, (short)0, acc[0][0], false, false);
        acc[0][1] = __builtin_amdgcn_wmma_f32_16x16x32_bf16(false, a0, false, b1, (short)0, acc[0][1], false, false);
        acc[1][0] = __builtin_amdgcn_wmma_f32_16x16x32_bf16(false, a1, false, b0, (short)0, acc[1][0], false, false);
        acc[1][1] = __builtin_amdgcn_wmma_f32_16x16x32_bf16(false, a1, false, b1, (short)0, acc[1][1], false, false);
    }

    // 32-row tiles never straddle a batch boundary (2048 % 32 == 0).
    const int bb    = row0 / S_;
    const int sbase = row0 % S_;

#pragma unroll
    for (int cc = 0; cc < 2; ++cc) {
        const int   n    = col0 + cc * 16 + bcol;
        const float bias = bqkv[n];

        if (n < 2 * D_) {
            // ---- Q or K: apply RoPE, store [B,H,S,HD] ----
            const int  nloc = (n < D_) ? n : (n - D_);
            const int  h    = nloc / HD_;
            const int  dd   = nloc % HD_;
            const float invf = __expf(-(float)(dd & ~1) * ROPE_LN_OVER_HD);
            __bf16* __restrict__ dst =
                ((n < D_) ? qb : kb) + ((size_t)bb * H_ + h) * S_ * HD_;
#pragma unroll
            for (int mm = 0; mm < 2; ++mm)
#pragma unroll
                for (int j = 0; j < 8; ++j) {
                    float v       = acc[mm][cc][j] + bias;
                    float partner = __shfl_xor(v, 1, 32);   // pair column (dd^1)
                    const int ss = sbase + mm * 16 + mhalf + j;
                    const float ang = (float)ss * invf;
                    const float cv  = __cosf(ang);
                    const float sv  = __sinf(ang);
                    // even dd: tr*cos - ti*sin ; odd dd: tr*sin + ti*cos
                    const float outv = ((dd & 1) == 0) ? (v * cv - partner * sv)
                                                       : (partner * sv + v * cv);
                    dst[(size_t)ss * HD_ + dd] = (__bf16)outv;
                }
        } else {
            // ---- V: store transposed [B,H,HD,S] ----
            const int nloc = n - 2 * D_;
            const int h    = nloc / HD_;
            const int dd   = nloc % HD_;
            __bf16* __restrict__ dst =
                vtb + (((size_t)bb * H_ + h) * HD_ + dd) * S_;
#pragma unroll
            for (int mm = 0; mm < 2; ++mm)
#pragma unroll
                for (int j = 0; j < 8; ++j)
                    dst[sbase + mm * 16 + mhalf + j] = (__bf16)(acc[mm][cc][j] + bias);
        }
    }
}

// ---------------------------------------------------------------------------
// Kernel 2: causal flash attention, one wave per (b, h, 16-query tile).
// Per 32-key step: 8 WMMAs for S = Q*K^T, online softmax, P through LDS
// (C-layout -> A-layout transpose), 8 WMMAs for O += P*V.
// All fragment loads are 16/32-byte vector loads (bf16, no conversion VALU).
// ---------------------------------------------------------------------------
__global__ __launch_bounds__(32)
void flash_attn_kernel(const __bf16* __restrict__ qb,
                       const __bf16* __restrict__ kb,
                       const __bf16* __restrict__ vtb,
                       __bf16* __restrict__ ob)
{
    __shared__ __bf16 Pbuf[16][32];

    const int lane  = threadIdx.x & 31;
    const int qt    = blockIdx.x;
    const int b     = qt / (H_ * (S_ / 16));
    const int rem   = qt % (H_ * (S_ / 16));
    const int h     = rem / (S_ / 16);
    const int qbase = (rem % (S_ / 16)) * 16;

    const size_t bh = (size_t)b * H_ + h;
    const __bf16* __restrict__ qptr = qb  + (bh * S_ + qbase) * HD_;
    const __bf16* __restrict__ kptr = kb  + bh * S_ * HD_;
    const __bf16* __restrict__ vptr = vtb + bh * HD_ * S_;

    const int arow  = lane & 15;
    const int aklo  = (lane < 16) ? 0 : 16;
    const int bcol  = lane & 15;
    const int bkoff = (lane < 16) ? 0 : 16;
    const int mhalf = (lane < 16) ? 0 : 8;

    // Q fragments: 4 chunks of 16x32 covering HD=128 (kept in registers).
    v16bf aq[4];
#pragma unroll
    for (int c = 0; c < 4; ++c)
        aq[c] = pack16(*(const v8bf*)(qptr + (size_t)arow * HD_ + c * 32 + aklo),
                       *(const v8bf*)(qptr + (size_t)arow * HD_ + c * 32 + 8 + aklo));

    const v8f vzero = {};
    v8f o[8];
#pragma unroll
    for (int nf = 0; nf < 8; ++nf) o[nf] = vzero;

    float mrow[8], lrow[8];
#pragma unroll
    for (int j = 0; j < 8; ++j) { mrow[j] = -__builtin_inff(); lrow[j] = 0.0f; }

    const float scale = 0.08838834764831845f;   // 1/sqrt(128)
    const int kend = qbase + 16;                // exclusive key bound (causal)

    for (int kk = 0; kk < kend; kk += 32) {
        // ---- scores: S[16x32] = Q[16x128] * K^T[128x32] ----
        v8f sc[2];
        sc[0] = vzero; sc[1] = vzero;
#pragma unroll
        for (int f = 0; f < 2; ++f) {
            const __bf16* __restrict__ kp =
                kptr + (size_t)(kk + f * 16 + bcol) * HD_;
#pragma unroll
            for (int c = 0; c < 4; ++c) {
                const v16bf bk = *(const v16bf*)(kp + c * 32 + bkoff);
                sc[f] = __builtin_amdgcn_wmma_f32_16x16x32_bf16(
                            false, aq[c], false, bk, (short)0, sc[f], false, false);
            }
        }

        // ---- scale + causal mask + online softmax (row stats per half-wave) ----
        float mnew[8];
#pragma unroll
        for (int j = 0; j < 8; ++j) {
            const int q = qbase + mhalf + j;
            float s0 = sc[0][j] * scale;
            float s1 = sc[1][j] * scale;
            if (kk + bcol      > q) s0 = -__builtin_inff();
            if (kk + 16 + bcol > q) s1 = -__builtin_inff();
            sc[0][j] = s0; sc[1][j] = s1;
            float rm = fmaxf(s0, s1);
            rm = fmaxf(rm, __shfl_xor(rm, 1, 32));
            rm = fmaxf(rm, __shfl_xor(rm, 2, 32));
            rm = fmaxf(rm, __shfl_xor(rm, 4, 32));
            rm = fmaxf(rm, __shfl_xor(rm, 8, 32));
            mnew[j] = fmaxf(mrow[j], rm);
        }
#pragma unroll
        for (int j = 0; j < 8; ++j) {
            const float alpha = __expf(mrow[j] - mnew[j]);
            const float p0 = __expf(sc[0][j] - mnew[j]);
            const float p1 = __expf(sc[1][j] - mnew[j]);
            float rs = p0 + p1;
            rs += __shfl_xor(rs, 1, 32);
            rs += __shfl_xor(rs, 2, 32);
            rs += __shfl_xor(rs, 4, 32);
            rs += __shfl_xor(rs, 8, 32);
            lrow[j] = lrow[j] * alpha + rs;
            mrow[j] = mnew[j];
#pragma unroll
            for (int nf = 0; nf < 8; ++nf) o[nf][j] *= alpha;
            Pbuf[mhalf + j][bcol]      = (__bf16)p0;   // C-layout -> LDS
            Pbuf[mhalf + j][16 + bcol] = (__bf16)p1;
        }
        __syncthreads();   // single-wave block: LDS write->read ordering

        // ---- P as A-matrix fragment (16x32) ----
        const v16bf ap = pack16(*(const v8bf*)(&Pbuf[arow][aklo]),
                                *(const v8bf*)(&Pbuf[arow][8 + aklo]));
        __syncthreads();   // reads done before next iteration overwrites

        // ---- O[16x128] += P[16x32] * V[32x128]  (V transposed in memory) ----
#pragma unroll
        for (int nf = 0; nf < 8; ++nf) {
            const v16bf bv =
                *(const v16bf*)(vptr + (size_t)(nf * 16 + bcol) * S_ + kk + bkoff);
            o[nf] = __builtin_amdgcn_wmma_f32_16x16x32_bf16(
                        false, ap, false, bv, (short)0, o[nf], false, false);
        }
    }

    // ---- normalize and store O as bf16 token-major [B,S,D] ----
#pragma unroll
    for (int j = 0; j < 8; ++j) {
        const float inv_l = 1.0f / lrow[j];
        const int   q     = qbase + mhalf + j;
        __bf16* __restrict__ orow = ob + ((size_t)b * S_ + q) * D_ + h * HD_;
#pragma unroll
        for (int nf = 0; nf < 8; ++nf)
            orow[nf * 16 + bcol] = (__bf16)(o[nf][j] * inv_l);
    }
}

// ---------------------------------------------------------------------------
// Kernel 3: out = O @ Wo^T + bo, fp32 output.
// One wave per 32x32 output block (2x2 WMMA register blocking).
// ---------------------------------------------------------------------------
__global__ __launch_bounds__(256)
void out_proj_kernel(const __bf16* __restrict__ ob,
                     const float* __restrict__ Wo,
                     const float* __restrict__ bo,
                     float* __restrict__ out)
{
    const int wave = blockIdx.x * 8 + (threadIdx.x >> 5);
    const int lane = threadIdx.x & 31;
    const int mt = wave % (NT_ / 32);
    const int nt = wave / (NT_ / 32);
    if (nt >= D_ / 32) return;

    const int arow  = lane & 15;
    const int aklo  = (lane < 16) ? 0 : 16;
    const int bcol  = lane & 15;
    const int bkoff = (lane < 16) ? 0 : 16;
    const int mhalf = (lane < 16) ? 0 : 8;

    const __bf16* __restrict__ or0 = ob + (size_t)(mt * 32 + arow) * D_;
    const __bf16* __restrict__ or1 = or0 + (size_t)16 * D_;
    const float*  __restrict__ wr0 = Wo + (size_t)(nt * 32 + bcol) * D_;
    const float*  __restrict__ wr1 = wr0 + (size_t)16 * D_;

    const v8f vzero = {};
    v8f acc[2][2];
    acc[0][0] = vzero; acc[0][1] = vzero; acc[1][0] = vzero; acc[1][1] = vzero;

    for (int k0 = 0; k0 < D_; k0 += 32) {
        v16bf a0 = pack16(*(const v8bf*)(or0 + k0 + aklo),
                          *(const v8bf*)(or0 + k0 + 8 + aklo));
        v16bf a1 = pack16(*(const v8bf*)(or1 + k0 + aklo),
                          *(const v8bf*)(or1 + k0 + 8 + aklo));
        v16bf b0 = cvt16(*(const v8f*)(wr0 + k0 + bkoff),
                         *(const v8f*)(wr0 + k0 + bkoff + 8));
        v16bf b1 = cvt16(*(const v8f*)(wr1 + k0 + bkoff),
                         *(const v8f*)(wr1 + k0 + bkoff + 8));
        acc[0][0] = __builtin_amdgcn_wmma_f32_16x16x32_bf16(false, a0, false, b0, (short)0, acc[0][0], false, false);
        acc[0][1] = __builtin_amdgcn_wmma_f32_16x16x32_bf16(false, a0, false, b1, (short)0, acc[0][1], false, false);
        acc[1][0] = __builtin_amdgcn_wmma_f32_16x16x32_bf16(false, a1, false, b0, (short)0, acc[1][0], false, false);
        acc[1][1] = __builtin_amdgcn_wmma_f32_16x16x32_bf16(false, a1, false, b1, (short)0, acc[1][1], false, false);
    }

#pragma unroll
    for (int cc = 0; cc < 2; ++cc) {
        const int   n    = nt * 32 + cc * 16 + bcol;
        const float bias = bo[n];
#pragma unroll
        for (int mm = 0; mm < 2; ++mm)
#pragma unroll
            for (int j = 0; j < 8; ++j)
                out[(size_t)(mt * 32 + mm * 16 + mhalf + j) * D_ + n] =
                    acc[mm][cc][j] + bias;
    }
}

// ---------------------------------------------------------------------------
// Launch: inputs are (x, Wqkv, bqkv, Wo, bo), all fp32. Output fp32 [B,S,D].
// Workspace layout (bf16): Q[BHSD] | K[BHSD] | Vt[BHSD] | O[BSD]  = ~134 MB.
// ---------------------------------------------------------------------------
extern "C" void kernel_launch(void* const* d_in, const int* in_sizes, int n_in,
                              void* d_out, int out_size, void* d_ws, size_t ws_size,
                              hipStream_t stream)
{
    (void)in_sizes; (void)n_in; (void)out_size; (void)ws_size;
    const float* x    = (const float*)d_in[0];
    const float* Wqkv = (const float*)d_in[1];
    const float* bqkv = (const float*)d_in[2];
    const float* Wo   = (const float*)d_in[3];
    const float* bo   = (const float*)d_in[4];
    float*       out  = (float*)d_out;

    const size_t BHSHD = (size_t)B_ * H_ * S_ * HD_;   // 16,777,216 elements
    char* ws = (char*)d_ws;
    __bf16* qb  = (__bf16*)(ws);
    __bf16* kb  = (__bf16*)(ws + 2 * BHSHD);
    __bf16* vtb = (__bf16*)(ws + 4 * BHSHD);
    __bf16* ob  = (__bf16*)(ws + 6 * BHSHD);

    // 1) QKV GEMM + bias + RoPE + pack: 256 x 192 32x32-blocks, 8 waves/block.
    qkv_rope_kernel<<<(NT_ / 32) * (QKVN / 32) / 8, 256, 0, stream>>>(
        x, Wqkv, bqkv, qb, kb, vtb);

    // 2) Flash attention: one wave per (b, h, 16-query tile).
    flash_attn_kernel<<<B_ * H_ * (S_ / 16), 32, 0, stream>>>(qb, kb, vtb, ob);

    // 3) Output projection: 256 x 64 32x32-blocks, 8 waves/block.
    out_proj_kernel<<<(NT_ / 32) * (D_ / 32) / 8, 256, 0, stream>>>(
        ob, Wo, bo, out);
}
